// QDSTModel_37538014167282
// MI455X (gfx1250) — compile-verified, matrix operand order checked by reading, not run on hardware
//
#include <hip/hip_runtime.h>
#include <cstddef>

// ---------------- model constants ----------------
#define S_   4096
#define D_   768
#define H_   12
#define DH_  64
#define F_   3072
#define G_   64
#define W_   256
#define C_   16
#define L_   6
#define NEG_   (-1.0e9f)
#define EPS_   (1e-5f)
#define SCALE_ (0.125f)   // 1/sqrt(64)

// ---------------- WMMA types ----------------
typedef __attribute__((ext_vector_type(16))) __bf16 v16bf;
typedef __attribute__((ext_vector_type(8)))  float  v8f;

union Frag  { unsigned u[8]; v16bf v; };
union Pack2 { __bf16 h[2]; unsigned u; };

// Pair index inside a 16x32 bf16 A/B fragment (ISA 7.12.2):
// half 0 holds K pairs {0..3, 8..11}, half 1 holds {4..7, 12..15}.
__device__ __forceinline__ int fpair(int p, int half) {
  return (p & 3) + (half << 2) + ((p >> 2) << 3);
}

__device__ __forceinline__ v8f wmma_bf16(v16bf a, v16bf b, v8f c) {
  return __builtin_amdgcn_wmma_f32_16x16x32_bf16(false, a, false, b, (short)0, c,
                                                 false, false);
}

// =====================================================================
// Weight convert + transpose:  out[n*K + k] = bf16(W[k*N + n])
// 32x32 tiles through LDS, coalesced on both global sides.
// =====================================================================
__global__ __launch_bounds__(256)
void wconv_kernel(const float* __restrict__ W, __bf16* __restrict__ out,
                  int K, int N)
{
  __shared__ __bf16 t[32][33];
  const int tid = threadIdx.x;
  const int n0 = blockIdx.x * 32, k0 = blockIdx.y * 32;
  for (int i = 0; i < 4; i++) {
    int u = tid + 256 * i;
    int k = u >> 5, n = u & 31;
    t[k][n] = (__bf16)W[(size_t)(k0 + k) * N + n0 + n];
  }
  __syncthreads();
  for (int i = 0; i < 4; i++) {
    int u = tid + 256 * i;
    int n = u >> 5, k = u & 31;
    out[(size_t)(n0 + n) * K + k0 + k] = t[k][n];
  }
}

// =====================================================================
// bf16 WMMA GEMM: out = act((A@W + bias) * alpha) + resid
//   A  : M x K bf16 row-major (M rows clamp-read; M % 16 == 0)
//   Bt : N x K bf16 (pre-transposed/converted weights)
//   block tile 128x128, 256 threads = 8 waves, each wave 32x64
//   N % 128 == 0, K % 32 == 0 assumed.
// act: 0 = none, 1 = exact GELU
// =====================================================================
#define LDS_STRIDE 40   // 80 B rows -> 16 B aligned for b128 staging

__global__ __launch_bounds__(256)
void gemm_bf16_kernel(const __bf16* __restrict__ A, const __bf16* __restrict__ Bt,
                      const float* __restrict__ bias, const float* __restrict__ resid,
                      float* __restrict__ outF, __bf16* __restrict__ outB,
                      int M, int K, int N, float alpha, int act)
{
  __shared__ __bf16 As[128 * LDS_STRIDE];
  __shared__ __bf16 Bs[128 * LDS_STRIDE];

  const int tid  = threadIdx.x;
  const int wave = tid >> 5;
  const int lane = tid & 31;
  const int half = lane >> 4;
  const int lr   = lane & 15;

  const int m0 = blockIdx.y * 128;
  const int n0 = blockIdx.x * 128;
  const int wm = (wave >> 1) * 32;   // wave row offset (4 wave-rows)
  const int wn = (wave & 1) * 64;    // wave col offset (2 wave-cols)

  v8f acc[2][4];
  for (int i = 0; i < 2; i++)
    for (int j = 0; j < 4; j++)
      acc[i][j] = (v8f){0.f,0.f,0.f,0.f,0.f,0.f,0.f,0.f};

  for (int k0 = 0; k0 < K; k0 += 32) {
    // prefetch next K tile; wave-uniform split (waves 0-3: A, waves 4-7: B)
    if (k0 + 32 < K) {
      if (tid < 128) {
        int pr = m0 + tid; if (pr > M - 1) pr = M - 1;
        __builtin_prefetch(A + (size_t)pr * K + k0 + 32, 0, 1);
      } else {
        __builtin_prefetch(Bt + (size_t)(n0 + (tid - 128)) * K + k0 + 32, 0, 1);
      }
    }
    // --- stage A + B tiles 128x32 with b128 loads (rows clamp to M-1) ---
    for (int i = 0; i < 2; i++) {
      int u = tid + 256 * i;            // 512 b128 loads per tile
      int row = u >> 2, seg = u & 3;
      int grow = m0 + row; if (grow > M - 1) grow = M - 1;
      uint4 av = *(const uint4*)(A  + (size_t)grow * K       + k0 + seg * 8);
      uint4 bv = *(const uint4*)(Bt + (size_t)(n0 + row) * K + k0 + seg * 8);
      *(uint4*)(&As[row * LDS_STRIDE + seg * 8]) = av;
      *(uint4*)(&Bs[row * LDS_STRIDE + seg * 8]) = bv;
    }
    __syncthreads();

    Frag af[2], bf[4];
    for (int t = 0; t < 2; t++)
      for (int p = 0; p < 8; p++) {
        int pp = fpair(p, half);
        af[t].u[p] = *(const unsigned*)(&As[(wm + t * 16 + lr) * LDS_STRIDE + pp * 2]);
      }
    for (int t = 0; t < 4; t++)
      for (int p = 0; p < 8; p++) {
        int pp = fpair(p, half);
        bf[t].u[p] = *(const unsigned*)(&Bs[(wn + t * 16 + lr) * LDS_STRIDE + pp * 2]);
      }
    for (int i = 0; i < 2; i++)
      for (int j = 0; j < 4; j++)
        acc[i][j] = wmma_bf16(af[i].v, bf[j].v, acc[i][j]);
    __syncthreads();
  }

  // --- epilogue (row guard is uniform per 16-row sub-tile since M%16==0) ---
  for (int i = 0; i < 2; i++) {
    if (m0 + wm + i * 16 >= M) continue;
    for (int j = 0; j < 4; j++)
      for (int e = 0; e < 8; e++) {
        int row = m0 + wm + i * 16 + e + 8 * half;   // C layout: M = e + 8*half
        int col = n0 + wn + j * 16 + lr;             // N = lane%16
        float r = (acc[i][j][e] + bias[col]) * alpha;
        if (act == 1) r = 0.5f * r * (1.0f + erff(r * 0.70710678118654752f));
        if (resid) r += resid[(size_t)row * N + col];
        if (outF)  outF[(size_t)row * N + col] = r;
        if (outB)  outB[(size_t)row * N + col] = (__bf16)r;
      }
  }
}

// =====================================================================
// Banded + global-key attention (one head, one chunk, 16-query tile).
// scores layout: [64 global keys | 768 band keys]  (832 total)
// kmap[col] = memory row of the key (-1 = invalid / masked)
// =====================================================================
#define SC_STRIDE 840  // 832 + pad

__global__ __launch_bounds__(128)
void band_attn_kernel(const __bf16* __restrict__ qb, const __bf16* __restrict__ kb,
                      const __bf16* __restrict__ vb, const int* __restrict__ sent_locs,
                      const float* __restrict__ tok_mask, float* __restrict__ attn)
{
  __shared__ float  sc[16 * SC_STRIDE];  // scores then probs
  __shared__ __bf16 Qs[16 * 68];
  __shared__ int    kmap[832];
  __shared__ float  red[16 * 8];

  const int tid  = threadIdx.x;
  const int wave = tid >> 5;
  const int lane = tid & 31;
  const int half = lane >> 4;
  const int lr   = lane & 15;

  const int qt = blockIdx.x;   // query tile within chunk (0..15)
  const int c  = blockIdx.y;   // chunk (0..15)
  const int hh = blockIdx.z;   // head  (0..11)

  // build key map (indices + range check + token mask baked in)
  for (int col = tid; col < 832; col += 128) {
    int key;
    if (col < G_) key = sent_locs[col];
    else {
      int kp = c * W_ - W_ + (col - G_);
      key = (kp >= 0 && kp < S_) ? kp : -1;
    }
    if (key >= 0 && !(tok_mask[key] > 0.f)) key = -1;
    kmap[col] = key;
  }

  // stage Q tile: 16 rows x 64 dims (q already scaled by 1/sqrt(DH))
  for (int i = 0; i < 4; i++) {
    int u = tid + 128 * i;            // 512 pair-loads
    int row = u >> 5, cp = u & 31;
    int sq = c * W_ + qt * 16 + row;
    unsigned val = *(const unsigned*)(qb + (size_t)sq * D_ + hh * DH_ + cp * 2);
    *(unsigned*)(&Qs[row * 68 + cp * 2]) = val;
  }
  __syncthreads();

  // Q fragments, one per 32-wide K half of DH=64
  Frag aq[2];
  for (int kh = 0; kh < 2; kh++)
    for (int p = 0; p < 8; p++) {
      int pp = fpair(p, half);
      aq[kh].u[p] = *(const unsigned*)(&Qs[lr * 68 + kh * 32 + pp * 2]);
    }

  // ---------- score phase: 52 key tiles of 16 ----------
  for (int t = wave; t < 52; t += 4) {
    int col = t * 16 + lr;                 // this lane's key column (0..831)
    int key = kmap[col];
    bool kvalid = (key >= 0);
    int krow = kvalid ? key : 0;
    Frag bk[2];
    for (int kh = 0; kh < 2; kh++)
      for (int p = 0; p < 8; p++) {
        int pp = fpair(p, half);
        bk[kh].u[p] = kvalid
          ? *(const unsigned*)(kb + (size_t)krow * D_ + hh * DH_ + kh * 32 + pp * 2)
          : 0u;
      }
    v8f a = (v8f){0.f,0.f,0.f,0.f,0.f,0.f,0.f,0.f};
    a = wmma_bf16(aq[0].v, bk[0].v, a);
    a = wmma_bf16(aq[1].v, bk[1].v, a);

    int ki = col - G_;                     // band key index (col >= 64)
    for (int e = 0; e < 8; e++) {
      int m  = e + 8 * half;
      int qi = qt * 16 + m;
      bool ok = kvalid;
      if (col >= G_) {
        int dlt = ki - W_ - qi;
        ok = ok && (dlt <= W_) && (dlt >= -W_);
      }
      sc[m * SC_STRIDE + col] = ok ? a[e] : NEG_;
    }
  }
  __syncthreads();

  // ---------- softmax over 832 keys, 8 threads per row ----------
  {
    int row = tid >> 3, part = tid & 7;
    float mx = -3.4e38f;
    for (int j = part; j < 832; j += 8) mx = fmaxf(mx, sc[row * SC_STRIDE + j]);
    red[row * 8 + part] = mx;
    __syncthreads();
    if (part == 0) {
      float m2 = red[row * 8];
      for (int i = 1; i < 8; i++) m2 = fmaxf(m2, red[row * 8 + i]);
      red[row * 8] = m2;
    }
    __syncthreads();
    float rmax = red[row * 8];
    float sum = 0.f;
    for (int j = part; j < 832; j += 8) {
      float e = __expf(sc[row * SC_STRIDE + j] - rmax);
      sc[row * SC_STRIDE + j] = e;
      sum += e;
    }
    __syncthreads();
    red[row * 8 + part] = sum;
    __syncthreads();
    if (part == 0) {
      float s2 = 0.f;
      for (int i = 0; i < 8; i++) s2 += red[row * 8 + i];
      red[row * 8] = s2;
    }
    __syncthreads();
    float inv = 1.0f / red[row * 8];
    for (int j = part; j < 832; j += 8) sc[row * SC_STRIDE + j] *= inv;
  }
  __syncthreads();

  // ---------- PV phase: out(16q x 64d) = P(16x832) @ V(832x64) ----------
  {
    const int dim = wave * 16 + lr;       // this lane's output dim (B fragment col)
    const size_t voff = (size_t)hh * DH_ + dim;
    v8f o = (v8f){0.f,0.f,0.f,0.f,0.f,0.f,0.f,0.f};
    for (int k0 = 0; k0 < 832; k0 += 32) {
      Frag ap, bv;
      for (int p = 0; p < 8; p++) {
        int pp = fpair(p, half);
        int cb = k0 + pp * 2;
        Pack2 pa;
        pa.h[0] = (__bf16)sc[lr * SC_STRIDE + cb];
        pa.h[1] = (__bf16)sc[lr * SC_STRIDE + cb + 1];
        ap.u[p] = pa.u;
        int ka = kmap[cb], kb2 = kmap[cb + 1];
        Pack2 pb;
        pb.h[0] = (ka  >= 0) ? vb[(size_t)ka  * D_ + voff] : (__bf16)0.0f;
        pb.h[1] = (kb2 >= 0) ? vb[(size_t)kb2 * D_ + voff] : (__bf16)0.0f;
        bv.u[p] = pb.u;
      }
      o = wmma_bf16(ap.v, bv.v, o);
    }
    for (int e = 0; e < 8; e++) {
      int m  = e + 8 * half;
      int sq = c * W_ + qt * 16 + m;
      attn[(size_t)sq * D_ + hh * DH_ + dim] = o[e];
    }
  }
}

// =====================================================================
// Global-token full attention (one head, one global token per block).
// Overwrites attn rows at sent_locs (runs after band kernel).
// =====================================================================
__global__ __launch_bounds__(256)
void global_attn_kernel(const __bf16* __restrict__ qgb, const __bf16* __restrict__ kgf,
                        const __bf16* __restrict__ vgf, const int* __restrict__ sent_locs,
                        const float* __restrict__ tok_mask, float* __restrict__ attn)
{
  __shared__ float sc[S_];
  __shared__ float qv[DH_];
  __shared__ float red[256];
  __shared__ float outp[4 * DH_];

  const int tid = threadIdx.x;
  const int g   = blockIdx.x;
  const int hh  = blockIdx.y;

  if (tid < DH_) qv[tid] = (float)qgb[(size_t)g * D_ + hh * DH_ + tid];
  __syncthreads();

  float mx = -3.4e38f;
  for (int s = tid; s < S_; s += 256) {
    const __bf16* kr = kgf + (size_t)s * D_ + hh * DH_;
    float dot = 0.f;
    for (int d = 0; d < DH_; d++) dot += qv[d] * (float)kr[d];
    if (!(tok_mask[s] > 0.f)) dot = NEG_;
    sc[s] = dot;
    mx = fmaxf(mx, dot);
  }
  red[tid] = mx; __syncthreads();
  for (int st = 128; st > 0; st >>= 1) {
    if (tid < st) red[tid] = fmaxf(red[tid], red[tid + st]);
    __syncthreads();
  }
  float rmax = red[0]; __syncthreads();

  float sum = 0.f;
  for (int s = tid; s < S_; s += 256) {
    float e = __expf(sc[s] - rmax);
    sc[s] = e; sum += e;
  }
  red[tid] = sum; __syncthreads();
  for (int st = 128; st > 0; st >>= 1) {
    if (tid < st) red[tid] += red[tid + st];
    __syncthreads();
  }
  float inv = 1.0f / red[0];

  int d = tid & 63, part = tid >> 6;
  float acc = 0.f;
  for (int s = part; s < S_; s += 4)
    acc += sc[s] * (float)vgf[(size_t)s * D_ + hh * DH_ + d];
  outp[part * DH_ + d] = acc;
  __syncthreads();
  if (tid < DH_) {
    float o = (outp[tid] + outp[DH_ + tid] + outp[2 * DH_ + tid] + outp[3 * DH_ + tid]) * inv;
    attn[(size_t)sent_locs[g] * D_ + hh * DH_ + tid] = o;
  }
}

// =====================================================================
// LayerNorm (one row per block) -> f32 + bf16 outputs
// =====================================================================
__global__ __launch_bounds__(256)
void ln_kernel(const float* __restrict__ in, const float* __restrict__ gs,
               const float* __restrict__ gb, float* __restrict__ outF,
               __bf16* __restrict__ outB)
{
  __shared__ float red[256];
  const int row = blockIdx.x, tid = threadIdx.x;
  const float* x = in + (size_t)row * D_;
  float s = 0.f;
  for (int j = tid; j < D_; j += 256) s += x[j];
  red[tid] = s; __syncthreads();
  for (int st = 128; st > 0; st >>= 1) { if (tid < st) red[tid] += red[tid + st]; __syncthreads(); }
  float mean = red[0] / (float)D_; __syncthreads();
  float v = 0.f;
  for (int j = tid; j < D_; j += 256) { float d = x[j] - mean; v += d * d; }
  red[tid] = v; __syncthreads();
  for (int st = 128; st > 0; st >>= 1) { if (tid < st) red[tid] += red[tid + st]; __syncthreads(); }
  float inv = rsqrtf(red[0] / (float)D_ + EPS_);
  for (int j = tid; j < D_; j += 256) {
    float y = (x[j] - mean) * inv * gs[j] + gb[j];
    outF[(size_t)row * D_ + j] = y;
    outB[(size_t)row * D_ + j] = (__bf16)y;
  }
}

// Fused embedding gather + LayerNorm
__global__ __launch_bounds__(256)
void embed_ln_kernel(const int* __restrict__ ids, const float* __restrict__ emb_w,
                     const float* __restrict__ emb_p, const float* __restrict__ gs,
                     const float* __restrict__ gb, float* __restrict__ outF,
                     __bf16* __restrict__ outB)
{
  __shared__ float xs[D_];
  __shared__ float red[256];
  const int row = blockIdx.x, tid = threadIdx.x;
  const int id = ids[row];
  float s = 0.f;
  for (int j = tid; j < D_; j += 256) {
    float e = emb_w[(size_t)id * D_ + j] + emb_p[(size_t)row * D_ + j];
    xs[j] = e; s += e;
  }
  red[tid] = s; __syncthreads();
  for (int st = 128; st > 0; st >>= 1) { if (tid < st) red[tid] += red[tid + st]; __syncthreads(); }
  float mean = red[0] / (float)D_; __syncthreads();
  float v = 0.f;
  for (int j = tid; j < D_; j += 256) { float d = xs[j] - mean; v += d * d; }
  red[tid] = v; __syncthreads();
  for (int st = 128; st > 0; st >>= 1) { if (tid < st) red[tid] += red[tid + st]; __syncthreads(); }
  float inv = rsqrtf(red[0] / (float)D_ + EPS_);
  for (int j = tid; j < D_; j += 256) {
    float y = (xs[j] - mean) * inv * gs[j] + gb[j];
    outF[(size_t)row * D_ + j] = y;
    outB[(size_t)row * D_ + j] = (__bf16)y;
  }
}

__global__ void gather_rows_kernel(const __bf16* __restrict__ hb,
                                   const int* __restrict__ locs,
                                   __bf16* __restrict__ out)
{
  int g = blockIdx.x, srow = locs[g];
  for (int j = threadIdx.x; j < D_; j += 256)
    out[(size_t)g * D_ + j] = hb[(size_t)srow * D_ + j];
}

__global__ void f2b_kernel(const float* __restrict__ in, __bf16* __restrict__ out, int n)
{
  int i = blockIdx.x * 256 + threadIdx.x;
  if (i < n) out[i] = (__bf16)in[i];
}

// pooled = tanh(h[0]@Wpool+b); hid = tanh(pooled@Whid+b); out = hid@Wfin+b
__global__ __launch_bounds__(256)
void head_kernel(const float* __restrict__ h, const float* __restrict__ Wpool,
                 const float* __restrict__ bpool, const float* __restrict__ Whid,
                 const float* __restrict__ bhid, const float* __restrict__ Wfin,
                 const float* __restrict__ bfin, float* __restrict__ out)
{
  __shared__ float pooled[D_];
  __shared__ float hid[D_];
  __shared__ float red[256];
  const int tid = threadIdx.x;
  for (int j = tid; j < D_; j += 256) {
    float a = 0.f;
    for (int d = 0; d < D_; d++) a += h[d] * Wpool[(size_t)d * D_ + j];
    pooled[j] = tanhf(a + bpool[j]);
  }
  __syncthreads();
  for (int j = tid; j < D_; j += 256) {
    float a = 0.f;
    for (int d = 0; d < D_; d++) a += pooled[d] * Whid[(size_t)d * D_ + j];
    hid[j] = tanhf(a + bhid[j]);
  }
  __syncthreads();
  float s = 0.f;
  for (int j = tid; j < D_; j += 256) s += hid[j] * Wfin[j];
  red[tid] = s; __syncthreads();
  for (int st = 128; st > 0; st >>= 1) { if (tid < st) red[tid] += red[tid + st]; __syncthreads(); }
  if (tid == 0) out[0] = red[0] + bfin[0];
}

// =====================================================================
// Host-side orchestration
// =====================================================================
extern "C" void kernel_launch(void* const* d_in, const int* in_sizes, int n_in,
                              void* d_out, int out_size, void* d_ws, size_t ws_size,
                              hipStream_t stream)
{
  (void)in_sizes; (void)n_in; (void)out_size; (void)ws_size;

  // inputs in setup_inputs() dict order (depth-first)
  const int*   input_ids = (const int*)  d_in[0];
  const float* tok_mask  = (const float*)d_in[1];
  const int*   sent_locs = (const int*)  d_in[2];
  // d_in[3] sent_mask unused
  const float* emb_w = (const float*)d_in[4];
  const float* emb_p = (const float*)d_in[5];
  const float* elns  = (const float*)d_in[6];
  const float* elnb  = (const float*)d_in[7];
  const float* Wq  = (const float*)d_in[8];   const float* bq  = (const float*)d_in[9];
  const float* Wk  = (const float*)d_in[10];  const float* bk  = (const float*)d_in[11];
  const float* Wv  = (const float*)d_in[12];  const float* bv  = (const float*)d_in[13];
  const float* Wqg = (const float*)d_in[14];  const float* bqg = (const float*)d_in[15];
  const float* Wkg = (const float*)d_in[16];  const float* bkg = (const float*)d_in[17];
  const float* Wvg = (const float*)d_in[18];  const float* bvg = (const float*)d_in[19];
  const float* Wo  = (const float*)d_in[20];  const float* bo  = (const float*)d_in[21];
  const float* ln1s = (const float*)d_in[22]; const float* ln1b = (const float*)d_in[23];
  const float* W1  = (const float*)d_in[24];  const float* b1  = (const float*)d_in[25];
  const float* W2  = (const float*)d_in[26];  const float* b2  = (const float*)d_in[27];
  const float* ln2s = (const float*)d_in[28]; const float* ln2b = (const float*)d_in[29];
  const float* Wpool = (const float*)d_in[30]; const float* bpool = (const float*)d_in[31];
  const float* Whid  = (const float*)d_in[32]; const float* bhid  = (const float*)d_in[33];
  const float* Wfin  = (const float*)d_in[34]; const float* bfin  = (const float*)d_in[35];

  // workspace carve-out (~131 MB)
  char* p = (char*)d_ws;
  auto alloc = [&](size_t bytes) {
    char* r = p;
    p += (bytes + 255) & ~(size_t)255;
    return r;
  };
  float*  h     = (float*) alloc((size_t)S_ * D_ * 4);
  __bf16* hb    = (__bf16*)alloc((size_t)S_ * D_ * 2);
  float*  h1    = (float*) alloc((size_t)S_ * D_ * 4);
  __bf16* h1b   = (__bf16*)alloc((size_t)S_ * D_ * 2);
  float*  pre   = (float*) alloc((size_t)S_ * D_ * 4);
  float*  attn  = (float*) alloc((size_t)S_ * D_ * 4);
  __bf16* attnb = (__bf16*)alloc((size_t)S_ * D_ * 2);
  __bf16* qbuf  = (__bf16*)alloc((size_t)S_ * D_ * 2);
  __bf16* kbuf  = (__bf16*)alloc((size_t)S_ * D_ * 2);
  __bf16* vbuf  = (__bf16*)alloc((size_t)S_ * D_ * 2);
  __bf16* kgfb  = (__bf16*)alloc((size_t)S_ * D_ * 2);
  __bf16* vgfb  = (__bf16*)alloc((size_t)S_ * D_ * 2);
  __bf16* ffb   = (__bf16*)alloc((size_t)S_ * F_ * 2);
  __bf16* hgb   = (__bf16*)alloc((size_t)G_ * D_ * 2);
  __bf16* qgb   = (__bf16*)alloc((size_t)G_ * D_ * 2);
  __bf16* wbT   = (__bf16*)alloc((size_t)F_ * D_ * 2);  // transposed bf16 weights

  // convert+transpose weights, then GEMM against bf16 Bt
  auto gemm = [&](const __bf16* A, const float* Wt, const float* bias,
                  const float* resid, float* oF, __bf16* oB,
                  int M, int K, int N, float alpha, int act) {
    wconv_kernel<<<dim3(N / 32, K / 32), 256, 0, stream>>>(Wt, wbT, K, N);
    dim3 grid(N / 128, (M + 127) / 128);
    gemm_bf16_kernel<<<grid, dim3(256), 0, stream>>>(A, wbT, bias, resid, oF, oB,
                                                     M, K, N, alpha, act);
  };

  // embedding + LN
  embed_ln_kernel<<<S_, 256, 0, stream>>>(input_ids, emb_w, emb_p, elns, elnb, h, hb);

  for (int l = 0; l < L_; l++) {
    const size_t dd = (size_t)D_ * D_, df = (size_t)D_ * F_;
    const float* Wq_l  = Wq  + l * dd; const float* bq_l  = bq  + (size_t)l * D_;
    const float* Wk_l  = Wk  + l * dd; const float* bk_l  = bk  + (size_t)l * D_;
    const float* Wv_l  = Wv  + l * dd; const float* bv_l  = bv  + (size_t)l * D_;
    const float* Wqg_l = Wqg + l * dd; const float* bqg_l = bqg + (size_t)l * D_;
    const float* Wkg_l = Wkg + l * dd; const float* bkg_l = bkg + (size_t)l * D_;
    const float* Wvg_l = Wvg + l * dd; const float* bvg_l = bvg + (size_t)l * D_;
    const float* Wo_l  = Wo  + l * dd; const float* bo_l  = bo  + (size_t)l * D_;
    const float* W1_l  = W1  + l * df; const float* b1_l  = b1  + (size_t)l * F_;
    const float* W2_l  = W2  + l * df; const float* b2_l  = b2  + (size_t)l * D_;
    const float* ln1s_l = ln1s + (size_t)l * D_; const float* ln1b_l = ln1b + (size_t)l * D_;
    const float* ln2s_l = ln2s + (size_t)l * D_; const float* ln2b_l = ln2b + (size_t)l * D_;

    // projections (q pre-scaled by 1/sqrt(DH))
    gemm(hb, Wq_l,  bq_l,  nullptr, nullptr, qbuf, S_, D_, D_, SCALE_, 0);
    gemm(hb, Wk_l,  bk_l,  nullptr, nullptr, kbuf, S_, D_, D_, 1.0f,   0);
    gemm(hb, Wv_l,  bv_l,  nullptr, nullptr, vbuf, S_, D_, D_, 1.0f,   0);
    gemm(hb, Wkg_l, bkg_l, nullptr, nullptr, kgfb, S_, D_, D_, 1.0f,   0);
    gemm(hb, Wvg_l, bvg_l, nullptr, nullptr, vgfb, S_, D_, D_, 1.0f,   0);

    // global-token queries
    gather_rows_kernel<<<G_, 256, 0, stream>>>(hb, sent_locs, hgb);
    gemm(hgb, Wqg_l, bqg_l, nullptr, nullptr, qgb, G_, D_, D_, SCALE_, 0);

    // attention
    band_attn_kernel<<<dim3(16, C_, H_), 128, 0, stream>>>(qbuf, kbuf, vbuf,
                                                           sent_locs, tok_mask, attn);
    global_attn_kernel<<<dim3(G_, H_), 256, 0, stream>>>(qgb, kgfb, vgfb,
                                                         sent_locs, tok_mask, attn);
    f2b_kernel<<<(S_ * D_) / 256, 256, 0, stream>>>(attn, attnb, S_ * D_);

    // output projection + residual, LN1
    gemm(attnb, Wo_l, bo_l, h, pre, nullptr, S_, D_, D_, 1.0f, 0);
    ln_kernel<<<S_, 256, 0, stream>>>(pre, ln1s_l, ln1b_l, h1, h1b);

    // FFN (exact GELU fused), residual, LN2 back into the stream
    gemm(h1b, W1_l, b1_l, nullptr, nullptr, ffb, S_, D_, F_, 1.0f, 1);
    gemm(ffb, W2_l, b2_l, h1, pre, nullptr, S_, F_, D_, 1.0f, 0);
    ln_kernel<<<S_, 256, 0, stream>>>(pre, ln2s_l, ln2b_l, h, hb);
  }

  head_kernel<<<1, 256, 0, stream>>>(h, Wpool, bpool, Whid, bhid, Wfin, bfin,
                                     (float*)d_out);
}
